// MultiHeadSelfAttention_69372311765400
// MI455X (gfx1250) — matrix operand order repaired and checked by reading, not compiled, not run on hardware
//
#include <hip/hip_runtime.h>
#include <hip/hip_bf16.h>

typedef __attribute__((ext_vector_type(16))) __bf16 v16bf;
typedef __attribute__((ext_vector_type(8)))  __bf16 v8bf;
typedef __attribute__((ext_vector_type(8)))  float  v8f;

#define DEV __device__ __forceinline__

static constexpr int B_  = 2;
static constexpr int S_  = 2048;
static constexpr int DM  = 1024;
static constexpr int HQ  = 16;
static constexpr int HKV = 4;
static constexpr int DQK = 64;

// Build a 16-element bf16 A-operand from two contiguous 8-element chunks.
// (A-matrix 16x32 bf16 layout used here: lane half h=0 holds K {0..15},
//  h=1 holds K {16..31}; caller passes the right chunk addresses.)
DEV v16bf load_a_pair(const __bf16* p0, const __bf16* p1) {
  v8bf lo = *(const v8bf*)p0;
  v8bf hi = *(const v8bf*)p1;
  v16bf a;
#pragma unroll
  for (int i = 0; i < 8; ++i) { a[i] = lo[i]; a[i + 8] = hi[i]; }
  return a;
}

DEV v8f wmma_bf16(v16bf a, v16bf b, v8f c) {
  return __builtin_amdgcn_wmma_f32_16x16x32_bf16(
      /*neg_a=*/false, a, /*neg_b=*/false, b,
      /*c_mod=*/(short)0, c, /*reuse_a=*/false, /*reuse_b=*/false);
}

// ---------------------------------------------------------------- convert
__global__ void cvt_bf16(const float* __restrict__ in, __bf16* __restrict__ out, int n) {
  int i = (blockIdx.x * blockDim.x + threadIdx.x) * 4;
  if (i + 3 < n) {
    float4 v = *(const float4*)(in + i);
    out[i + 0] = (__bf16)v.x;
    out[i + 1] = (__bf16)v.y;
    out[i + 2] = (__bf16)v.z;
    out[i + 3] = (__bf16)v.w;
  }
}

// ------------------------------------------------------- QKV proj + RoPE
// One wave per 16x64 output strip (4 N-tiles share each A operand).
// 24 strips per row tile: 16 (Q: 1024 cols) + 4 (K: 256) + 4 (V: 256).
__global__ void __launch_bounds__(128) qkv_rope_kernel(
    const __bf16* __restrict__ xb,
    const __bf16* __restrict__ wq, const __bf16* __restrict__ wk,
    const __bf16* __restrict__ wv,
    __bf16* __restrict__ Qr, __bf16* __restrict__ Kr, __bf16* __restrict__ Vt) {
  const int lane = threadIdx.x & 31;
  const int wave = (blockIdx.x * blockDim.x + threadIdx.x) >> 5;
  const int ST = 24;
  const int rt = wave / ST;              // 0..255  (B*S/16 row tiles)
  const int st = wave % ST;
  const int lm = lane & 15;
  const bool hic = lane >= 16;
  const int ha = hic ? 8 : 0;            // accumulator row half offset
  const int hb = hic ? 16 : 0;           // B-operand half offset

  const __bf16* w;
  int n0, kind;
  if (st < 16)      { w = wq; n0 = st * 64;        kind = 0; }
  else if (st < 20) { w = wk; n0 = (st - 16) * 64; kind = 1; }
  else              { w = wv; n0 = (st - 20) * 64; kind = 2; }

  const __bf16* arow = xb + (size_t)(rt * 16 + lm) * DM;
  const __bf16* brow0 = w + (size_t)(n0 + lm) * DM;

  v8f c[4] = {{}, {}, {}, {}};
#pragma unroll 2
  for (int kb = 0; kb < DM; kb += 32) {
    __builtin_prefetch(arow + kb + 256, 0, 3);
    __builtin_prefetch(brow0 + kb + 256, 0, 3);
    v16bf a = load_a_pair(arow + kb + hb, arow + kb + hb + 8);
#pragma unroll
    for (int j = 0; j < 4; ++j) {
      v16bf b = *(const v16bf*)(brow0 + (size_t)j * 16 * DM + kb + hb);
      c[j] = wmma_bf16(a, b, c[j]);
    }
  }

  const int g0 = rt * 16;
  const int bb = g0 / S_;
  const int s0 = g0 % S_;

#pragma unroll
  for (int j = 0; j < 4; ++j) {
    const int col = n0 + j * 16 + lm;
    if (kind == 2) {
      // V: store transposed  Vt[(b*HKV+h)*64 + d][s]
      const int h = col >> 6, d = col & 63;
      __bf16* vp = Vt + ((size_t)(bb * HKV + h) * DQK + d) * S_;
#pragma unroll
      for (int r = 0; r < 8; ++r)
        vp[s0 + r + ha] = (__bf16)c[j][r];
    } else {
      // RoPE: pair columns (2j, 2j+1) via lane^1 exchange
      const int d = col & 63;
      const float inv = __expf(-(float)(d & ~1) * 0.14391156507f); // ln(1e4)/64
      const float sgn = (d & 1) ? 1.0f : -1.0f;
#pragma unroll
      for (int r = 0; r < 8; ++r) {
        float v = c[j][r];
        float p = __shfl_xor(v, 1, 32);
        float ang = (float)(s0 + r + ha) * inv;
        float cs = __cosf(ang), sn = __sinf(ang);
        c[j][r] = v * cs + sgn * p * sn;
      }
      const int h = col >> 6, dd = col & 63;
      __bf16* op = (kind == 0)
          ? Qr + (size_t)(bb * HQ + h) * S_ * DQK
          : Kr + (size_t)(bb * HKV + h) * S_ * DQK;
#pragma unroll
      for (int r = 0; r < 8; ++r)
        op[(size_t)(s0 + r + ha) * DQK + dd] = (__bf16)c[j][r];
    }
  }
}

// ----------------------------------------------------- flash attention
// One wave per 16-query tile.  Scores computed transposed (lane = query).
__global__ void __launch_bounds__(128) attn_kernel(
    const __bf16* __restrict__ Qr, const __bf16* __restrict__ Kr,
    const __bf16* __restrict__ Vt, __bf16* __restrict__ attnb) {
  const int lane = threadIdx.x & 31;
  const int wave = (blockIdx.x * blockDim.x + threadIdx.x) >> 5;
  const int QT = S_ / 16;
  const int bb = wave / (HQ * QT);
  const int h  = (wave / QT) % HQ;
  const int q0 = (wave % QT) * 16;
  const int hkv = h / (HQ / HKV);
  const int lm = lane & 15;
  const bool hic = lane >= 16;
  const int ha = hic ? 8 : 0;
  const int hb = hic ? 16 : 0;

  const __bf16* Qp = Qr + (size_t)(bb * HQ + h) * S_ * DQK;
  const __bf16* Kp = Kr + (size_t)(bb * HKV + hkv) * S_ * DQK;
  const __bf16* Vp = Vt + (size_t)(bb * HKV + hkv) * DQK * S_;

  // Resident Q B-operands (B-matrix layout: half picks K 0..15 / 16..31)
  v16bf bq0 = *(const v16bf*)(Qp + (size_t)(q0 + lm) * DQK + hb);
  v16bf bq1 = *(const v16bf*)(Qp + (size_t)(q0 + lm) * DQK + 32 + hb);

  v8f o[4] = {{}, {}, {}, {}};
  float m_run = -3.0e38f, l_run = 0.0f;
  const int qg = q0 + lm;

  for (int k0 = 0; k0 < q0 + 16; k0 += 32) {
    // S^T tiles (M = key, N = query): tile0 keys k0..k0+15, tile1 +16
    const __bf16* kr0 = Kp + (size_t)(k0 + lm) * DQK;
    const __bf16* kr1 = kr0 + 16 * DQK;
    v8f s0t = {}, s1t = {};
    {
      v16bf ka = load_a_pair(kr0 + hb, kr0 + hb + 8);
      v16bf kb = load_a_pair(kr0 + 32 + hb, kr0 + 40 + hb);
      s0t = wmma_bf16(ka, bq0, s0t);
      s0t = wmma_bf16(kb, bq1, s0t);
    }
    {
      v16bf ka = load_a_pair(kr1 + hb, kr1 + hb + 8);
      v16bf kb = load_a_pair(kr1 + 32 + hb, kr1 + 40 + hb);
      s1t = wmma_bf16(ka, bq0, s1t);
      s1t = wmma_bf16(kb, bq1, s1t);
    }

    // clamp(+-80) / sqrt(64) scale / causal mask, online softmax stats
    float p0[8], p1[8];
    float mloc = -3.0e38f;
#pragma unroll
    for (int r = 0; r < 8; ++r) {
      int kg0 = k0 + r + ha;
      int kg1 = kg0 + 16;
      float v0 = fminf(fmaxf(s0t[r], -80.0f), 80.0f) * 0.125f;
      float v1 = fminf(fmaxf(s1t[r], -80.0f), 80.0f) * 0.125f;
      if (kg0 > qg) v0 = -1.0e30f;
      if (kg1 > qg) v1 = -1.0e30f;
      p0[r] = v0; p1[r] = v1;
      mloc = fmaxf(mloc, fmaxf(v0, v1));
    }
    mloc = fmaxf(mloc, __shfl_xor(mloc, 16, 32));
    float m_new = fmaxf(m_run, mloc);
    float scale = __expf(m_run - m_new);
    float rsum = 0.0f;
#pragma unroll
    for (int r = 0; r < 8; ++r) {
      p0[r] = __expf(p0[r] - m_new);
      p1[r] = __expf(p1[r] - m_new);
      rsum += p0[r] + p1[r];
    }
    rsum += __shfl_xor(rsum, 16, 32);
    l_run = l_run * scale + rsum;
    m_run = m_new;
#pragma unroll
    for (int r = 0; r < 8; ++r) {
      o[0][r] *= scale; o[1][r] *= scale; o[2][r] *= scale; o[3][r] *= scale;
    }

    // Assemble P^T B-operand (K = 32 keys, N = 16 queries); lane^16 swap
    v16bf pb;
#pragma unroll
    for (int r = 0; r < 8; ++r) {
      float t0 = __shfl_xor(p0[r], 16, 32);
      float t1 = __shfl_xor(p1[r], 16, 32);
      pb[r]     = hic ? (__bf16)t1    : (__bf16)p0[r];
      pb[r + 8] = hic ? (__bf16)p1[r] : (__bf16)t0;
    }

    // O^T += V^T(d x 32keys) @ P^T(32keys x 16q); Vt is d-major -> contig
#pragma unroll
    for (int dt = 0; dt < 4; ++dt) {
      const __bf16* vr = Vp + (size_t)(dt * 16 + lm) * S_ + k0;
      v16bf va = load_a_pair(vr + hb, vr + hb + 8);
      o[dt] = wmma_bf16(va, pb, o[dt]);
    }
  }

  const float inv_l = 1.0f / l_run;
  __bf16* op = attnb + (size_t)(bb * S_ + q0 + lm) * DM + h * DQK;
#pragma unroll
  for (int dt = 0; dt < 4; ++dt)
#pragma unroll
    for (int r = 0; r < 8; ++r)
      op[dt * 16 + r + ha] = (__bf16)(o[dt][r] * inv_l);
}

// ------------------------------------------------------ output projection
// One wave per 16x64 output strip (4 N-tiles share each A operand).
__global__ void __launch_bounds__(128) out_proj_kernel(
    const __bf16* __restrict__ attnb, const __bf16* __restrict__ wo,
    float* __restrict__ out) {
  const int lane = threadIdx.x & 31;
  const int wave = (blockIdx.x * blockDim.x + threadIdx.x) >> 5;
  const int rt = wave / 16;
  const int st = wave % 16;
  const int n0 = st * 64;
  const int lm = lane & 15;
  const bool hic = lane >= 16;
  const int ha = hic ? 8 : 0;
  const int hb = hic ? 16 : 0;

  const __bf16* arow = attnb + (size_t)(rt * 16 + lm) * DM;
  const __bf16* brow0 = wo + (size_t)(n0 + lm) * DM;

  v8f c[4] = {{}, {}, {}, {}};
#pragma unroll 2
  for (int kb = 0; kb < DM; kb += 32) {
    __builtin_prefetch(arow + kb + 256, 0, 3);
    __builtin_prefetch(brow0 + kb + 256, 0, 3);
    v16bf a = load_a_pair(arow + kb + hb, arow + kb + hb + 8);
#pragma unroll
    for (int j = 0; j < 4; ++j) {
      v16bf b = *(const v16bf*)(brow0 + (size_t)j * 16 * DM + kb + hb);
      c[j] = wmma_bf16(a, b, c[j]);
    }
  }
#pragma unroll
  for (int j = 0; j < 4; ++j)
#pragma unroll
    for (int r = 0; r < 8; ++r)
      out[(size_t)(rt * 16 + r + ha) * DM + n0 + j * 16 + lm] = c[j][r];
}

// ----------------------------------------------------------------- launch
extern "C" void kernel_launch(void* const* d_in, const int* in_sizes, int n_in,
                              void* d_out, int out_size, void* d_ws, size_t ws_size,
                              hipStream_t stream) {
  const float* x  = (const float*)d_in[0];
  const float* Wq = (const float*)d_in[1];
  const float* Wk = (const float*)d_in[2];
  const float* Wv = (const float*)d_in[3];
  const float* Wo = (const float*)d_in[4];
  float* out = (float*)d_out;

  char* ws = (char*)d_ws;
  size_t off = 0;
  auto alloc = [&](size_t bytes) {
    char* p = ws + off;
    off = (off + bytes + 255) & ~(size_t)255;
    return p;
  };
  __bf16* xb  = (__bf16*)alloc((size_t)B_ * S_ * DM * 2);
  __bf16* wqb = (__bf16*)alloc((size_t)DM * DM * 2);
  __bf16* wkb = (__bf16*)alloc((size_t)HKV * DQK * DM * 2);
  __bf16* wvb = (__bf16*)alloc((size_t)HKV * DQK * DM * 2);
  __bf16* wob = (__bf16*)alloc((size_t)DM * DM * 2);
  __bf16* Qr  = (__bf16*)alloc((size_t)B_ * HQ * S_ * DQK * 2);
  __bf16* Kr  = (__bf16*)alloc((size_t)B_ * HKV * S_ * DQK * 2);
  __bf16* Vt  = (__bf16*)alloc((size_t)B_ * HKV * DQK * S_ * 2);
  __bf16* ab  = (__bf16*)alloc((size_t)B_ * S_ * DM * 2);

  auto cvt = [&](const float* in, __bf16* o, int n) {
    int nv = n / 4;
    cvt_bf16<<<(nv + 255) / 256, 256, 0, stream>>>(in, o, n);
  };
  cvt(x,  xb,  B_ * S_ * DM);
  cvt(Wq, wqb, DM * DM);
  cvt(Wk, wkb, HKV * DQK * DM);
  cvt(Wv, wvb, HKV * DQK * DM);
  cvt(Wo, wob, DM * DM);

  // QKV + RoPE: 256 row-tiles * 24 strips, 4 waves/block
  qkv_rope_kernel<<<(256 * 24) / 4, 128, 0, stream>>>(xb, wqb, wkb, wvb, Qr, Kr, Vt);

  // Attention: B*HQ*(S/16) = 4096 waves
  attn_kernel<<<(B_ * HQ * (S_ / 16)) / 4, 128, 0, stream>>>(Qr, Kr, Vt, ab);

  // Output projection: 256 row-tiles * 16 strips
  out_proj_kernel<<<(256 * 16) / 4, 128, 0, stream>>>(ab, wob, out);
}
